// MultiHeadNetwork_6305011991033
// MI455X (gfx1250) — matrix operand order repaired and checked by reading, not compiled, not run on hardware
//
#include <hip/hip_runtime.h>
#include <hip/hip_bf16.h>
#include <stdint.h>

#define BATCH 4096
#define FEAT 2048
#define NUM_TASKS 50
#define WIDTH 2048
#define HEAD_DIM 256
#define XIN (FEAT + NUM_TASKS)   /* 2098 */
#define K0P 2112                 /* K padded to multiple of 32 for layer 0 */

typedef unsigned short u16;
typedef unsigned int   u32;

typedef __attribute__((ext_vector_type(16))) __bf16 v16bf;
typedef __attribute__((ext_vector_type(8)))  float  v8f;

__device__ __forceinline__ u16 f2bf(float f) {
  u32 u = __float_as_uint(f);
  u32 r = u + 0x7FFFu + ((u >> 16) & 1u);   // round-to-nearest-even
  return (u16)(r >> 16);
}

union FragBF {
  uint4 q[2];
  v16bf v;
};

// Generic LDS pointer -> 32-bit DS byte offset (low 32 bits per aperture rules)
__device__ __forceinline__ unsigned lds_off32(const void* p) {
  return (unsigned)(uintptr_t)p;
}

// CDNA5 async direct-to-LDS 16B copy (per-lane), tracked by ASYNCcnt.
__device__ __forceinline__ void async_copy16(unsigned lds_byte_off, const u16* gptr) {
  asm volatile("global_load_async_to_lds_b128 %0, %1, off"
               :
               : "v"(lds_byte_off), "v"((unsigned long long)(uintptr_t)gptr)
               : "memory");
}

__device__ __forceinline__ void wait_async0() {
  asm volatile("s_wait_asynccnt 0x0" ::: "memory");
}

// ---------------------------------------------------------------------------
// counts[] zeroing (must happen every call; harness does not re-poison)
// ---------------------------------------------------------------------------
__global__ void zero_counts(int* counts) {
  if (threadIdx.x < NUM_TASKS) counts[threadIdx.x] = 0;
}

// ---------------------------------------------------------------------------
// Per-row: argmax of one-hot tail -> bucket row by task; convert x -> bf16
// padded to K0P with zeros.
// ---------------------------------------------------------------------------
__global__ __launch_bounds__(256)
void prep_rows(const float* __restrict__ x, u16* __restrict__ xb,
               int* __restrict__ counts, int* __restrict__ rowlist) {
  const int b = blockIdx.x;
  const float* xr = x + (size_t)b * XIN;
  u16* xbr = xb + (size_t)b * K0P;
  for (int j = threadIdx.x; j < K0P; j += 256) {
    float v = (j < XIN) ? xr[j] : 0.0f;
    xbr[j] = f2bf(v);
  }
  if (threadIdx.x == 0) {
    const float* oh = xr + FEAT;
    int best = 0; float bv = oh[0];
    for (int i = 1; i < NUM_TASKS; i++) {
      float v = oh[i];
      if (v > bv) { bv = v; best = i; }
    }
    int pos = atomicAdd(&counts[best], 1);
    rowlist[best * BATCH + pos] = b;
  }
}

// ---------------------------------------------------------------------------
// Transpose + f32->bf16 convert: W[K][N] (row-major f32) -> Wt[N][Kp] bf16,
// zero-padding K..Kp. blockIdx.z selects a slice (for per-task head weights).
// ---------------------------------------------------------------------------
__global__ __launch_bounds__(256)
void transpose_bf16(const float* __restrict__ W, u16* __restrict__ Wt,
                    int K, int N, int Kp) {
  W  += (size_t)blockIdx.z * K * N;
  Wt += (size_t)blockIdx.z * N * Kp;
  __shared__ float tile[32][33];
  const int n0 = blockIdx.x * 32;
  const int k0 = blockIdx.y * 32;
  const int tx = threadIdx.x & 31;
  const int ty = threadIdx.x >> 5;   // 0..7
  #pragma unroll
  for (int i = 0; i < 32; i += 8) {
    int k = k0 + ty + i, n = n0 + tx;
    float v = (k < K && n < N) ? W[(size_t)k * N + n] : 0.0f;
    tile[ty + i][tx] = v;
  }
  __syncthreads();
  #pragma unroll
  for (int i = 0; i < 32; i += 8) {
    int n = n0 + ty + i, k = k0 + tx;
    if (n < N && k < Kp) Wt[(size_t)n * Kp + k] = f2bf(tile[tx][ty + i]);
  }
}

// ---------------------------------------------------------------------------
// C[M][N] = relu(A[M][K] @ Bt[N][K]^T + bias), bf16 in / bf16 out, f32 acc.
// Workgroup tile 128x128, 8 waves (4 in M x 2 in N), each wave 32x64 via
// 2x4 WMMA f32_16x16x32_bf16 accumulators.
// K staged in steps of 32 through DOUBLE-BUFFERED LDS filled with
// global_load_async_to_lds_b128 (ASYNCcnt) -> one barrier per K-step.
// ---------------------------------------------------------------------------
#define LPITCH 40                 /* LDS row pitch in halves (80 B) */
#define AREG  (128 * LPITCH)      /* halves per A (or B) region     */

__global__ __launch_bounds__(256)
void gemm_relu_bf16(const u16* __restrict__ A, int lda,
                    const u16* __restrict__ Bt, int ldb,
                    const float* __restrict__ bias,
                    u16* __restrict__ C, int ldc,
                    int K) {
  __shared__ __align__(16) u16 sbuf[2][2 * AREG];   // [buf][ A | B ]

  const int tid  = threadIdx.x;
  const int lane = tid & 31;
  const int wave = tid >> 5;
  const int wm   = wave & 3;     // M group (32 rows each)
  const int wn   = wave >> 2;    // N group (64 cols each)
  const int tileM = blockIdx.y * 128;
  const int tileN = blockIdx.x * 128;

  v8f zero8;
  #pragma unroll
  for (int e = 0; e < 8; e++) zero8[e] = 0.0f;
  v8f acc[2][4];
  #pragma unroll
  for (int i = 0; i < 2; i++)
    #pragma unroll
    for (int j = 0; j < 4; j++) acc[i][j] = zero8;

  // staging map: chunk c -> row c>>2, 16B part c&3 ; thread does c=tid, tid+256
  const int r0 = tid >> 2;
  const int p0 = tid & 3;
  const u16* gA0 = A  + (size_t)(tileM + r0)      * lda + p0 * 8;
  const u16* gA1 = A  + (size_t)(tileM + r0 + 64) * lda + p0 * 8;
  const u16* gB0 = Bt + (size_t)(tileN + r0)      * ldb + p0 * 8;
  const u16* gB1 = Bt + (size_t)(tileN + r0 + 64) * ldb + p0 * 8;

  const unsigned bufbase0 = lds_off32(&sbuf[0][0]);
  const unsigned bufbase1 = lds_off32(&sbuf[1][0]);
  const unsigned offA0 = (unsigned)((r0 * LPITCH + p0 * 8) * 2);
  const unsigned offA1 = (unsigned)(((r0 + 64) * LPITCH + p0 * 8) * 2);
  const unsigned offB0 = (unsigned)((AREG + r0 * LPITCH + p0 * 8) * 2);
  const unsigned offB1 = (unsigned)((AREG + (r0 + 64) * LPITCH + p0 * 8) * 2);

  const int mrow   = wm * 32 + (lane & 15);
  const int nrow   = wn * 64 + (lane & 15);
  const int klaneA = (lane & 16) ? 8  : 0;   // A: lane<16 K0-7/16-23, lane>=16 K8-15/24-31
  const int klaneB = (lane & 16) ? 8  : 0;   // B: lane<16 K0-7/16-23, lane>=16 K8-15/24-31

  // prologue: fill buffer 0 with the kk=0 tile
  {
    async_copy16(bufbase0 + offA0, gA0);
    async_copy16(bufbase0 + offA1, gA1);
    async_copy16(bufbase0 + offB0, gB0);
    async_copy16(bufbase0 + offB1, gB1);
  }

  int cur = 0;
  for (int kk = 0; kk < K; kk += 32) {
    wait_async0();        // my async writes for buffer `cur` have landed
    __syncthreads();      // everyone's landed; everyone done reading cur^1

    const int kn = kk + 32;
    if (kn < K) {
      const unsigned nb = cur ? bufbase0 : bufbase1;
      async_copy16(nb + offA0, gA0 + kn);
      async_copy16(nb + offA1, gA1 + kn);
      async_copy16(nb + offB0, gB0 + kn);
      async_copy16(nb + offB1, gB1 + kn);
    }

    const u16* sA = &sbuf[cur][0];
    const u16* sB = &sbuf[cur][AREG];

    FragBF af[2], bfrag[4];
    #pragma unroll
    for (int mf = 0; mf < 2; mf++) {
      const u16* p = sA + (mrow + mf * 16) * LPITCH + klaneA;
      af[mf].q[0] = *(const uint4*)(p);        // K 0-7  (or 8-15)
      af[mf].q[1] = *(const uint4*)(p + 16);   // K 16-23 (or 24-31)
    }
    #pragma unroll
    for (int nf = 0; nf < 4; nf++) {
      const u16* p = sB + (nrow + nf * 16) * LPITCH + klaneB;
      bfrag[nf].q[0] = *(const uint4*)(p);     // K 0-7  (or 8-15)
      bfrag[nf].q[1] = *(const uint4*)(p + 16); // K 16-23 (or 24-31)
    }
    #pragma unroll
    for (int mf = 0; mf < 2; mf++)
      #pragma unroll
      for (int nf = 0; nf < 4; nf++)
        acc[mf][nf] = __builtin_amdgcn_wmma_f32_16x16x32_bf16(
            false, af[mf].v, false, bfrag[nf].v,
            (short)0, acc[mf][nf], false, false);

    cur ^= 1;
  }

  // epilogue: bias + relu + bf16 store (C layout: VGPR i -> M=i / M=8+i)
  #pragma unroll
  for (int mf = 0; mf < 2; mf++) {
    #pragma unroll
    for (int nf = 0; nf < 4; nf++) {
      int n = tileN + wn * 64 + nf * 16 + (lane & 15);
      float bv = bias[n];
      #pragma unroll
      for (int i = 0; i < 8; i++) {
        int m = tileM + wm * 32 + mf * 16 + ((lane & 16) ? (8 + i) : i);
        float v = acc[mf][nf][i] + bv;
        v = v > 0.0f ? v : 0.0f;
        C[(size_t)m * ldc + n] = f2bf(v);
      }
    }
  }
}

// ---------------------------------------------------------------------------
// Gathered heads: block = (task, 16-row tile of that task's bucket).
// 8 waves cover HEAD_DIM=256 (32 cols each, 2 WMMA accumulators per wave).
// ---------------------------------------------------------------------------
__global__ __launch_bounds__(256)
void head_kernel(const u16* __restrict__ h,        // [BATCH][WIDTH] bf16
                 const u16* __restrict__ headWt,   // [T][HEAD_DIM][WIDTH] bf16
                 const float* __restrict__ head_b, // [T][HEAD_DIM]
                 const int* __restrict__ counts,
                 const int* __restrict__ rowlist,  // [T][BATCH]
                 float* __restrict__ out) {        // [BATCH][HEAD_DIM]
  const int t     = blockIdx.y;
  const int mtile = blockIdx.x;
  const int cnt   = counts[t];
  if (mtile * 16 >= cnt) return;

  __shared__ int rows_s[16];
  const int tid = threadIdx.x;
  if (tid < 16) {
    int slot = mtile * 16 + tid;
    rows_s[tid] = (slot < cnt) ? rowlist[t * BATCH + slot] : -1;
  }
  __syncthreads();

  const int lane = tid & 31;
  const int wave = tid >> 5;                       // N offset = wave*32
  const int myrow = rows_s[lane & 15];
  const u16* arow = h + (size_t)(myrow < 0 ? 0 : myrow) * WIDTH;
  const u16* Bt   = headWt + (size_t)t * HEAD_DIM * WIDTH;
  const int klaneA = (lane & 16) ? 8  : 0;
  const int klaneB = (lane & 16) ? 8  : 0;

  v8f zero8;
  #pragma unroll
  for (int e = 0; e < 8; e++) zero8[e] = 0.0f;
  v8f acc[2] = {zero8, zero8};

  const u16* pb0base = Bt + (size_t)(wave * 32 + (lane & 15)) * WIDTH + klaneB;
  const u16* pb1base = pb0base + (size_t)16 * WIDTH;

  for (int kk = 0; kk < WIDTH; kk += 32) {
    FragBF a, b0, b1;
    const u16* pa = arow + kk + klaneA;
    a.q[0] = *(const uint4*)(pa);
    a.q[1] = *(const uint4*)(pa + 16);
    const u16* pb0 = pb0base + kk;
    b0.q[0] = *(const uint4*)(pb0);
    b0.q[1] = *(const uint4*)(pb0 + 16);
    const u16* pb1 = pb1base + kk;
    b1.q[0] = *(const uint4*)(pb1);
    b1.q[1] = *(const uint4*)(pb1 + 16);
    acc[0] = __builtin_amdgcn_wmma_f32_16x16x32_bf16(
        false, a.v, false, b0.v, (short)0, acc[0], false, false);
    acc[1] = __builtin_amdgcn_wmma_f32_16x16x32_bf16(
        false, a.v, false, b1.v, (short)0, acc[1], false, false);
  }

  #pragma unroll
  for (int nf = 0; nf < 2; nf++) {
    int n = wave * 32 + nf * 16 + (lane & 15);
    float bv = head_b[t * HEAD_DIM + n];
    #pragma unroll
    for (int i = 0; i < 8; i++) {
      int mm = (lane & 16) ? (8 + i) : i;
      int rr = rows_s[mm];
      if (rr >= 0) out[(size_t)rr * HEAD_DIM + n] = acc[nf][i] + bv;
    }
  }
}

// ---------------------------------------------------------------------------
extern "C" void kernel_launch(void* const* d_in, const int* in_sizes, int n_in,
                              void* d_out, int out_size, void* d_ws, size_t ws_size,
                              hipStream_t stream) {
  (void)in_sizes; (void)n_in; (void)out_size; (void)ws_size;
  const float* x     = (const float*)d_in[0];
  const float* W0    = (const float*)d_in[1];
  const float* b0    = (const float*)d_in[2];
  const float* W1    = (const float*)d_in[3];
  const float* b1    = (const float*)d_in[4];
  const float* W2    = (const float*)d_in[5];
  const float* b2    = (const float*)d_in[6];
  const float* headW = (const float*)d_in[7];
  const float* headb = (const float*)d_in[8];
  float* out = (float*)d_out;

  char* ws = (char*)d_ws;
  size_t off = 0;
  auto alloc = [&](size_t bytes) {
    char* p = ws + off;
    off = (off + bytes + 255) & ~(size_t)255;
    return p;
  };
  int* counts  = (int*)alloc(NUM_TASKS * sizeof(int));
  int* rowlist = (int*)alloc((size_t)NUM_TASKS * BATCH * sizeof(int));
  u16* xb      = (u16*)alloc((size_t)BATCH * K0P * sizeof(u16));
  u16* wbuf    = (u16*)alloc((size_t)WIDTH * K0P * sizeof(u16));      // reused per layer
  u16* h1      = (u16*)alloc((size_t)BATCH * WIDTH * sizeof(u16));
  u16* h2      = (u16*)alloc((size_t)BATCH * WIDTH * sizeof(u16));
  u16* hwt     = (u16*)alloc((size_t)NUM_TASKS * HEAD_DIM * WIDTH * sizeof(u16));

  zero_counts<<<dim3(1), dim3(64), 0, stream>>>(counts);
  prep_rows<<<dim3(BATCH), dim3(256), 0, stream>>>(x, xb, counts, rowlist);

  // head weights: [T][2048][256] f32 -> [T][256][2048] bf16
  transpose_bf16<<<dim3(HEAD_DIM / 32, WIDTH / 32, NUM_TASKS), dim3(256), 0, stream>>>(
      headW, hwt, WIDTH, HEAD_DIM, WIDTH);

  // layer 0: [4096 x 2112] @ [2112 x 2048]
  transpose_bf16<<<dim3(WIDTH / 32, K0P / 32, 1), dim3(256), 0, stream>>>(
      W0, wbuf, XIN, WIDTH, K0P);
  gemm_relu_bf16<<<dim3(WIDTH / 128, BATCH / 128), dim3(256), 0, stream>>>(
      xb, K0P, wbuf, K0P, b0, h1, WIDTH, K0P);

  // layer 1
  transpose_bf16<<<dim3(WIDTH / 32, WIDTH / 32, 1), dim3(256), 0, stream>>>(
      W1, wbuf, WIDTH, WIDTH, WIDTH);
  gemm_relu_bf16<<<dim3(WIDTH / 128, BATCH / 128), dim3(256), 0, stream>>>(
      h1, WIDTH, wbuf, WIDTH, b1, h2, WIDTH, WIDTH);

  // layer 2
  transpose_bf16<<<dim3(WIDTH / 32, WIDTH / 32, 1), dim3(256), 0, stream>>>(
      W2, wbuf, WIDTH, WIDTH, WIDTH);
  gemm_relu_bf16<<<dim3(WIDTH / 128, BATCH / 128), dim3(256), 0, stream>>>(
      h2, WIDTH, wbuf, WIDTH, b2, h1, WIDTH, WIDTH);

  // gathered per-task heads
  head_kernel<<<dim3(BATCH / 16, NUM_TASKS), dim3(256), 0, stream>>>(
      h1, hwt, headb, counts, rowlist, out);
}
